// LoRAAQExpert_11579231830502
// MI455X (gfx1250) — compile-verified, matrix operand order checked by reading, not run on hardware
//
#include <hip/hip_runtime.h>
#include <hip/hip_bf16.h>
#include <stdint.h>

// Problem constants (match reference)
#define HID    4096
#define INTER  11008
#define MTOT   8192          // B*S
#define LORA_R 128
#define NCODES 65536
#define LSCALE 2.0f          // lora_alpha / r = 256/128

typedef __attribute__((ext_vector_type(16))) __bf16 bf16x16;
typedef __attribute__((ext_vector_type(8)))  float  floatx8;

// ---------------- async global -> LDS (gfx1250) with safe fallback ----------
#if defined(__has_builtin)
#if __has_builtin(__builtin_amdgcn_global_load_async_to_lds_b128)
#define HAVE_ASYNC_LDS 1
#endif
#endif

typedef int v4i __attribute__((vector_size(16)));
typedef __attribute__((address_space(1))) v4i* gv4_ptr;
typedef __attribute__((address_space(3))) v4i* lv4_ptr;

__device__ __forceinline__ void copy16B_g2lds(const void* g, void* l) {
#ifdef HAVE_ASYNC_LDS
  __builtin_amdgcn_global_load_async_to_lds_b128(
      (gv4_ptr)(uintptr_t)g, (lv4_ptr)(uint32_t)(uintptr_t)l, 0, 0);
#else
  *(uint4*)l = *(const uint4*)g;
#endif
}

// stage 32B (16 bf16) per thread
__device__ __forceinline__ void stage32B(const __bf16* src, __bf16* dst) {
  copy16B_g2lds(src, dst);
  copy16B_g2lds(src + 8, dst + 8);
}

__device__ __forceinline__ void async_wait_all() {
#ifdef HAVE_ASYNC_LDS
  asm volatile("s_wait_asynccnt 0" ::: "memory");
#endif
}

// ---------------- WMMA helpers ----------------------------------------------
__device__ __forceinline__ floatx8 wmma_bf16(bf16x16 a, bf16x16 b, floatx8 c) {
  return __builtin_amdgcn_wmma_f32_16x16x32_bf16(false, a, false, b, (short)0, c,
                                                 false, false);
}

// A/B fragment from bf16 LDS tile stored row-major [rows][32].
// Lane l<16: row=row0+l, K 0..7 & 16..23 ; lane l>=16: row=row0+l-16, K 8..15 & 24..31.
union FragU { bf16x16 v; uint4 q[2]; };

__device__ __forceinline__ bf16x16 frag_bf16(const __bf16* base, int row0, int lane) {
  const int r  = row0 + (lane & 15);
  const int kh = (lane >> 4) * 8;
  const __bf16* p = base + r * 32 + kh;
  FragU f;
  f.q[0] = *(const uint4*)(p);
  f.q[1] = *(const uint4*)(p + 16);
  return f.v;
}

// AQ dequant of one group of 8 -> packed bf16 (uint4), scale folded in.
__device__ __forceinline__ uint4 dequant8v(int2 idx, const float* __restrict__ cb,
                                           const float* __restrict__ sc) {
  const float4* c0 = (const float4*)(cb + (size_t)idx.x * 8);
  const float4* c1 = (const float4*)(cb + ((size_t)NCODES + (size_t)idx.y) * 8);
  float4 a0 = c0[0], a1 = c0[1];
  float4 b0 = c1[0], b1 = c1[1];
  union { __bf16 h[8]; uint4 q; } o;
  o.h[0] = (__bf16)((a0.x + b0.x) * sc[0]);
  o.h[1] = (__bf16)((a0.y + b0.y) * sc[1]);
  o.h[2] = (__bf16)((a0.z + b0.z) * sc[2]);
  o.h[3] = (__bf16)((a0.w + b0.w) * sc[3]);
  o.h[4] = (__bf16)((a1.x + b1.x) * sc[4]);
  o.h[5] = (__bf16)((a1.y + b1.y) * sc[5]);
  o.h[6] = (__bf16)((a1.z + b1.z) * sc[6]);
  o.h[7] = (__bf16)((a1.w + b1.w) * sc[7]);
  return o.q;
}

// =============================================================================
// Kernel 0: fp32 -> bf16 bulk convert (8 elems / thread, exact division)
// =============================================================================
__global__ __launch_bounds__(256) void cvt_bf16_kernel(
    const float* __restrict__ in, __bf16* __restrict__ out) {
  const size_t i = (size_t)blockIdx.x * 256 + threadIdx.x;
  const float4* s = (const float4*)in + i * 2;
  float4 a = s[0], b = s[1];
  union { __bf16 h[8]; uint4 q; } o;
  o.h[0] = (__bf16)a.x; o.h[1] = (__bf16)a.y;
  o.h[2] = (__bf16)a.z; o.h[3] = (__bf16)a.w;
  o.h[4] = (__bf16)b.x; o.h[5] = (__bf16)b.y;
  o.h[6] = (__bf16)b.z; o.h[7] = (__bf16)b.w;
  ((uint4*)out)[i] = o.q;
}

// =============================================================================
// Kernel 1: fused gate/up AQ GEMM + silu*mul -> q_mid (bf16 [M, INTER]).
// 256 thr (8 waves), C tile 128x64, wave grid 4x2 (32x32 each).
// Double-buffered LDS; async copy + dequant for step k+1 overlap WMMA on k.
// =============================================================================
__global__ __launch_bounds__(256) void gateup_kernel(
    const __bf16* __restrict__ xb,
    const int* __restrict__ gidx, const float* __restrict__ gcb,
    const float* __restrict__ gsc,
    const int* __restrict__ uidx, const float* __restrict__ ucb,
    const float* __restrict__ usc,
    __bf16* __restrict__ qmid) {
  __shared__ __bf16 xs[2][128 * 32];
  __shared__ __bf16 wg[2][64 * 32];
  __shared__ __bf16 wu[2][64 * 32];

  const int tid  = threadIdx.x;
  const int lane = tid & 31;
  const int wave = tid >> 5;
  const int m0 = blockIdx.x * 128;
  const int n0 = blockIdx.y * 64;
  const int mw = (wave >> 1) * 32;
  const int nw = (wave & 1) * 32;

  const floatx8 zero = {0.f, 0.f, 0.f, 0.f, 0.f, 0.f, 0.f, 0.f};
  floatx8 accg[2][2] = {{zero, zero}, {zero, zero}};
  floatx8 accu[2][2] = {{zero, zero}, {zero, zero}};

  // staging / dequant thread roles
  const int sr = tid >> 1, sc = (tid & 1) * 16;          // x tile: 32B/thread
  const __bf16* xrow = xb + (size_t)(m0 + sr) * HID + sc;
  __bf16* xdst0 = &xs[0][sr * 32 + sc];
  __bf16* xdst1 = &xs[1][sr * 32 + sc];
  const int drow = tid >> 2, dg = tid & 3;               // weight tile group
  const size_t ibase = ((size_t)(n0 + drow) * (HID / 8) + dg) * 2;
  const int woff = drow * 32 + dg * 8;

  // ---- prologue: stage k0 = 0 ----
  stage32B(xrow, xdst0);
  {
    const int2 gi = *(const int2*)(gidx + ibase);
    const int2 ui = *(const int2*)(uidx + ibase);
    *(uint4*)&wg[0][woff] = dequant8v(gi, gcb, gsc + dg * 8);
    *(uint4*)&wu[0][woff] = dequant8v(ui, ucb, usc + dg * 8);
  }
  async_wait_all();
  __syncthreads();

  for (int k0 = 0; k0 < HID; k0 += 32) {
    const int cur = (k0 >> 5) & 1, nxt = cur ^ 1;
    const bool more = (k0 + 32) < HID;
    uint4 wgv = {}, wuv = {};
    if (more) {
      // issue async x copy + codebook gathers for next step
      stage32B(xrow + k0 + 32, nxt ? xdst1 : xdst0);
      const size_t ib = ibase + (size_t)((k0 + 32) >> 3) * 2;
      const int2 gi = *(const int2*)(gidx + ib);
      const int2 ui = *(const int2*)(uidx + ib);
      wgv = dequant8v(gi, gcb, gsc + (k0 + 32) + dg * 8);
      wuv = dequant8v(ui, ucb, usc + (k0 + 32) + dg * 8);
      __builtin_prefetch(gidx + ib + 8, 0, 0);   // 2 steps ahead
      __builtin_prefetch(uidx + ib + 8, 0, 0);
    }

    // compute on current buffers
    bf16x16 a0 = frag_bf16(xs[cur], mw, lane);
    bf16x16 a1 = frag_bf16(xs[cur], mw + 16, lane);
    bf16x16 g0 = frag_bf16(wg[cur], nw, lane);
    bf16x16 g1 = frag_bf16(wg[cur], nw + 16, lane);
    bf16x16 u0 = frag_bf16(wu[cur], nw, lane);
    bf16x16 u1 = frag_bf16(wu[cur], nw + 16, lane);

    accg[0][0] = wmma_bf16(a0, g0, accg[0][0]);
    accg[0][1] = wmma_bf16(a0, g1, accg[0][1]);
    accg[1][0] = wmma_bf16(a1, g0, accg[1][0]);
    accg[1][1] = wmma_bf16(a1, g1, accg[1][1]);
    accu[0][0] = wmma_bf16(a0, u0, accu[0][0]);
    accu[0][1] = wmma_bf16(a0, u1, accu[0][1]);
    accu[1][0] = wmma_bf16(a1, u0, accu[1][0]);
    accu[1][1] = wmma_bf16(a1, u1, accu[1][1]);

    if (more) {
      *(uint4*)&wg[nxt][woff] = wgv;
      *(uint4*)&wu[nxt][woff] = wuv;
      async_wait_all();
    }
    __syncthreads();
  }

  // epilogue: clip/nan, silu(gate)*up, store bf16
  const int colL = lane & 15;
  const int rowH = (lane >> 4) * 8;
#pragma unroll
  for (int mi = 0; mi < 2; ++mi)
#pragma unroll
    for (int ni = 0; ni < 2; ++ni)
#pragma unroll
      for (int v = 0; v < 8; ++v) {
        float g = accg[mi][ni][v];
        float u = accu[mi][ni][v];
        float gc = (g != g) ? 0.f : fminf(fmaxf(g, -50.f), 50.f);
        float uc = (u != u) ? 0.f : fminf(fmaxf(u, -50.f), 50.f);
        float s = gc / (1.f + __expf(-gc));
        const int row = m0 + mw + mi * 16 + v + rowH;
        const int col = n0 + nw + ni * 16 + colL;
        qmid[(size_t)row * INTER + col] = (__bf16)(s * uc);
      }
}

// =============================================================================
// Kernel 2: down AQ GEMM; out += 0.01 * postproc(q_mid @ Wd'^T).
// Same double-buffered pipeline (LoRA already in d_out).
// =============================================================================
__global__ __launch_bounds__(256) void down_kernel(
    const __bf16* __restrict__ qmid,
    const int* __restrict__ didx, const float* __restrict__ dcb,
    const float* __restrict__ dsc,
    float* __restrict__ out) {
  __shared__ __bf16 as_[2][128 * 32];
  __shared__ __bf16 wd[2][64 * 32];

  const int tid  = threadIdx.x;
  const int lane = tid & 31;
  const int wave = tid >> 5;
  const int m0 = blockIdx.x * 128;
  const int n0 = blockIdx.y * 64;     // over HID
  const int mw = (wave >> 1) * 32;
  const int nw = (wave & 1) * 32;

  const floatx8 zero = {0.f, 0.f, 0.f, 0.f, 0.f, 0.f, 0.f, 0.f};
  floatx8 acc[2][2] = {{zero, zero}, {zero, zero}};

  const int sr = tid >> 1, sc = (tid & 1) * 16;
  const __bf16* arow = qmid + (size_t)(m0 + sr) * INTER + sc;
  __bf16* adst0 = &as_[0][sr * 32 + sc];
  __bf16* adst1 = &as_[1][sr * 32 + sc];
  const int drow = tid >> 2, dg = tid & 3;
  const size_t ibase = ((size_t)(n0 + drow) * (INTER / 8) + dg) * 2;
  const int woff = drow * 32 + dg * 8;

  // prologue
  stage32B(arow, adst0);
  {
    const int2 di = *(const int2*)(didx + ibase);
    *(uint4*)&wd[0][woff] = dequant8v(di, dcb, dsc + dg * 8);
  }
  async_wait_all();
  __syncthreads();

  for (int k0 = 0; k0 < INTER; k0 += 32) {
    const int cur = (k0 >> 5) & 1, nxt = cur ^ 1;
    const bool more = (k0 + 32) < INTER;
    uint4 wdv = {};
    if (more) {
      stage32B(arow + k0 + 32, nxt ? adst1 : adst0);
      const size_t ib = ibase + (size_t)((k0 + 32) >> 3) * 2;
      const int2 di = *(const int2*)(didx + ib);
      wdv = dequant8v(di, dcb, dsc + (k0 + 32) + dg * 8);
      __builtin_prefetch(didx + ib + 8, 0, 0);
    }

    bf16x16 a0 = frag_bf16(as_[cur], mw, lane);
    bf16x16 a1 = frag_bf16(as_[cur], mw + 16, lane);
    bf16x16 b0 = frag_bf16(wd[cur], nw, lane);
    bf16x16 b1 = frag_bf16(wd[cur], nw + 16, lane);

    acc[0][0] = wmma_bf16(a0, b0, acc[0][0]);
    acc[0][1] = wmma_bf16(a0, b1, acc[0][1]);
    acc[1][0] = wmma_bf16(a1, b0, acc[1][0]);
    acc[1][1] = wmma_bf16(a1, b1, acc[1][1]);

    if (more) {
      *(uint4*)&wd[nxt][woff] = wdv;
      async_wait_all();
    }
    __syncthreads();
  }

  const int colL = lane & 15;
  const int rowH = (lane >> 4) * 8;
  const float inf = __builtin_inff();
#pragma unroll
  for (int mi = 0; mi < 2; ++mi)
#pragma unroll
    for (int ni = 0; ni < 2; ++ni)
#pragma unroll
      for (int v = 0; v < 8; ++v) {
        float q = acc[mi][ni][v];
        if (q != q)            q = 0.f;
        else if (q == inf)     q = 10.f;
        else if (q == -inf)    q = -10.f;
        else                   q = fminf(fmaxf(q, -50.f), 50.f);
        const int row = m0 + mw + mi * 16 + v + rowH;
        const int col = n0 + nw + ni * 16 + colL;
        const size_t o = (size_t)row * HID + col;
        out[o] = out[o] + 0.01f * q;
      }
}

// =============================================================================
// Kernel 3: t = x @ lora_A^T  (bf16 result [M, R])
// =============================================================================
__global__ __launch_bounds__(256) void lora_t_kernel(
    const __bf16* __restrict__ xb, const float* __restrict__ la,
    __bf16* __restrict__ t) {
  __shared__ __bf16 xs[128 * 32];
  __shared__ __bf16 wa[64 * 32];

  const int tid  = threadIdx.x;
  const int lane = tid & 31;
  const int wave = tid >> 5;
  const int m0 = blockIdx.x * 128;
  const int n0 = blockIdx.y * 64;     // over R=128
  const int mw = (wave >> 1) * 32;
  const int nw = (wave & 1) * 32;

  const floatx8 zero = {0.f, 0.f, 0.f, 0.f, 0.f, 0.f, 0.f, 0.f};
  floatx8 acc[2][2] = {{zero, zero}, {zero, zero}};

  for (int k0 = 0; k0 < HID; k0 += 32) {
    {  // stage x tile (bf16, async)
      const int r = tid >> 1, c = (tid & 1) * 16;
      stage32B(xb + (size_t)(m0 + r) * HID + k0 + c, xs + r * 32 + c);
    }
    {  // stage lora_A tile [64 x 32] fp32 -> bf16
      const int r = tid >> 2, c = (tid & 3) * 8;
      const float* src = la + (size_t)(n0 + r) * HID + k0 + c;
      union { __bf16 h[8]; uint4 q; } o;
#pragma unroll
      for (int i = 0; i < 8; ++i) o.h[i] = (__bf16)src[i];
      *(uint4*)(wa + r * 32 + c) = o.q;
    }
    async_wait_all();
    __syncthreads();

    bf16x16 a0 = frag_bf16(xs, mw, lane);
    bf16x16 a1 = frag_bf16(xs, mw + 16, lane);
    bf16x16 b0 = frag_bf16(wa, nw, lane);
    bf16x16 b1 = frag_bf16(wa, nw + 16, lane);

    acc[0][0] = wmma_bf16(a0, b0, acc[0][0]);
    acc[0][1] = wmma_bf16(a0, b1, acc[0][1]);
    acc[1][0] = wmma_bf16(a1, b0, acc[1][0]);
    acc[1][1] = wmma_bf16(a1, b1, acc[1][1]);

    __syncthreads();
  }

  const int colL = lane & 15;
  const int rowH = (lane >> 4) * 8;
#pragma unroll
  for (int mi = 0; mi < 2; ++mi)
#pragma unroll
    for (int ni = 0; ni < 2; ++ni)
#pragma unroll
      for (int v = 0; v < 8; ++v) {
        const int row = m0 + mw + mi * 16 + v + rowH;
        const int col = n0 + nw + ni * 16 + colL;
        t[(size_t)row * LORA_R + col] = (__bf16)acc[mi][ni][v];
      }
}

// =============================================================================
// Kernel 4: out = LSCALE * (t @ lora_B^T)   (writes d_out; runs before down)
// =============================================================================
__global__ __launch_bounds__(256) void lora_out_kernel(
    const __bf16* __restrict__ t, const float* __restrict__ lb,
    float* __restrict__ out) {
  __shared__ __bf16 ts[128 * 32];
  __shared__ __bf16 wb[64 * 32];

  const int tid  = threadIdx.x;
  const int lane = tid & 31;
  const int wave = tid >> 5;
  const int m0 = blockIdx.x * 128;
  const int n0 = blockIdx.y * 64;     // over HID
  const int mw = (wave >> 1) * 32;
  const int nw = (wave & 1) * 32;

  const floatx8 zero = {0.f, 0.f, 0.f, 0.f, 0.f, 0.f, 0.f, 0.f};
  floatx8 acc[2][2] = {{zero, zero}, {zero, zero}};

  for (int k0 = 0; k0 < LORA_R; k0 += 32) {
    {  // stage t tile [128 x 32] bf16 (pitch 128), async
      const int r = tid >> 1, c = (tid & 1) * 16;
      stage32B(t + (size_t)(m0 + r) * LORA_R + k0 + c, ts + r * 32 + c);
    }
    {  // stage lora_B tile [64 x 32] fp32 -> bf16 (pitch 128)
      const int r = tid >> 2, c = (tid & 3) * 8;
      const float* src = lb + (size_t)(n0 + r) * LORA_R + k0 + c;
      union { __bf16 h[8]; uint4 q; } o;
#pragma unroll
      for (int i = 0; i < 8; ++i) o.h[i] = (__bf16)src[i];
      *(uint4*)(wb + r * 32 + c) = o.q;
    }
    async_wait_all();
    __syncthreads();

    bf16x16 a0 = frag_bf16(ts, mw, lane);
    bf16x16 a1 = frag_bf16(ts, mw + 16, lane);
    bf16x16 b0 = frag_bf16(wb, nw, lane);
    bf16x16 b1 = frag_bf16(wb, nw + 16, lane);

    acc[0][0] = wmma_bf16(a0, b0, acc[0][0]);
    acc[0][1] = wmma_bf16(a0, b1, acc[0][1]);
    acc[1][0] = wmma_bf16(a1, b0, acc[1][0]);
    acc[1][1] = wmma_bf16(a1, b1, acc[1][1]);

    __syncthreads();
  }

  const int colL = lane & 15;
  const int rowH = (lane >> 4) * 8;
#pragma unroll
  for (int mi = 0; mi < 2; ++mi)
#pragma unroll
    for (int ni = 0; ni < 2; ++ni)
#pragma unroll
      for (int v = 0; v < 8; ++v) {
        const int row = m0 + mw + mi * 16 + v + rowH;
        const int col = n0 + nw + ni * 16 + colL;
        out[(size_t)row * HID + col] = LSCALE * acc[mi][ni][v];
      }
}

// =============================================================================
extern "C" void kernel_launch(void* const* d_in, const int* in_sizes, int n_in,
                              void* d_out, int out_size, void* d_ws,
                              size_t ws_size, hipStream_t stream) {
  (void)in_sizes; (void)n_in; (void)out_size; (void)ws_size;
  const float* x    = (const float*)d_in[0];
  const int*   gidx = (const int*)d_in[1];
  const float* gcb  = (const float*)d_in[2];
  const float* gsc  = (const float*)d_in[3];
  const int*   uidx = (const int*)d_in[4];
  const float* ucb  = (const float*)d_in[5];
  const float* usc  = (const float*)d_in[6];
  const int*   didx = (const int*)d_in[7];
  const float* dcb  = (const float*)d_in[8];
  const float* dsc  = (const float*)d_in[9];
  const float* lA   = (const float*)d_in[10];
  const float* lB   = (const float*)d_in[11];
  float* out = (float*)d_out;

  // workspace layout: q_mid bf16 [M,INTER] | x bf16 [M,HID] | t bf16 [M,R]
  char* ws = (char*)d_ws;
  __bf16* qmid = (__bf16*)ws;
  __bf16* xb   = (__bf16*)(ws + (size_t)MTOT * INTER * sizeof(short));
  __bf16* t    = (__bf16*)(ws + (size_t)MTOT * (INTER + HID) * sizeof(short));

  dim3 blk(256);
  cvt_bf16_kernel<<<dim3((MTOT * (size_t)HID) / (256 * 8)), blk, 0, stream>>>(x, xb);
  lora_t_kernel  <<<dim3(MTOT / 128, LORA_R / 64), blk, 0, stream>>>(xb, lA, t);
  lora_out_kernel<<<dim3(MTOT / 128, HID / 64),   blk, 0, stream>>>(t, lB, out);
  gateup_kernel  <<<dim3(MTOT / 128, INTER / 64), blk, 0, stream>>>(
      xb, gidx, gcb, gsc, uidx, ucb, usc, qmid);
  down_kernel    <<<dim3(MTOT / 128, HID / 64),   blk, 0, stream>>>(
      qmid, didx, dcb, dsc, out);
}
